// Patch_GCN_cls_3624952398640
// MI455X (gfx1250) — compile-verified, matrix operand order checked by reading, not run on hardware
//
#include <hip/hip_runtime.h>
#include <hip/hip_bf16.h>

// ---------------------------------------------------------------------------
// Types for CDNA5 WMMA (wave32): v16bf A/B fragments, v8f accumulator.
// ---------------------------------------------------------------------------
typedef __attribute__((ext_vector_type(16))) __bf16 bf16x16;
typedef __attribute__((ext_vector_type(8)))  __bf16 bf16x8;
typedef __attribute__((ext_vector_type(8)))  float  f32x8;

__device__ __forceinline__ __bf16 f2bf(float f) {
  unsigned u = __float_as_uint(f);
  u += 0x7FFFu + ((u >> 16) & 1u);           // round-to-nearest-even
  unsigned short s = (unsigned short)(u >> 16);
  __bf16 b;
  __builtin_memcpy(&b, &s, 2);
  return b;
}

// Convert float4 -> 4 consecutive bf16 (packed conversion when available).
__device__ __forceinline__ void cvt4_bf16(__bf16* dp, float4 v) {
#if __has_builtin(__builtin_amdgcn_cvt_pk_bf16_f32)
  auto p0 = __builtin_amdgcn_cvt_pk_bf16_f32(v.x, v.y);
  auto p1 = __builtin_amdgcn_cvt_pk_bf16_f32(v.z, v.w);
  __builtin_memcpy(dp,     &p0, 4);
  __builtin_memcpy(dp + 2, &p1, 4);
#else
  dp[0] = f2bf(v.x); dp[1] = f2bf(v.y); dp[2] = f2bf(v.z); dp[3] = f2bf(v.w);
#endif
}

// ---------------------------------------------------------------------------
// Pack f32 weight [K,N] row-major into bf16 WMMA B-fragment layout:
// tiles [nblk][kblk], 32 lanes x 16 bf16 per tile.
// Lane L (<16): N=L,   K = {0..7,16..23}+32*kblk ; lane>=16: K = {8..15,24..31}.
// Fragment element j: K-offset = (lane>>4)*8 + (j<8 ? j : j+8).
// ---------------------------------------------------------------------------
__global__ void pack_weights_k(const float* __restrict__ W, __bf16* __restrict__ Wp,
                               int K, int N) {
  int idx = blockIdx.x * 256 + threadIdx.x;
  if (idx >= K * N) return;
  int j    = idx & 15;
  int lane = (idx >> 4) & 31;
  int t    = idx >> 9;
  int kblks = K >> 5;
  int kblk = t % kblks;
  int nblk = t / kblks;
  int kk = ((lane >> 4) << 3) + (j < 8 ? j : j + 8);
  int k  = kblk * 32 + kk;
  int n  = nblk * 16 + (lane & 15);
  Wp[idx] = f2bf(W[(size_t)k * N + n]);
}

// ---------------------------------------------------------------------------
// WMMA GEMM: C[:, col_off:col_off+N] = act(A[M,K] @ W[K,N] + bias)
// 128 threads = 4 waves; block tile = 16 rows x 128 cols.
// Each wave owns two 16x16 N-tiles (dual accumulators) so one A fragment
// feeds two v_wmma per K-step. A tile (16x32 f32->bf16) staged to LDS with
// double buffering: one barrier per K-step, stage(k+1) overlaps wmma(k).
// ---------------------------------------------------------------------------
__global__ void __launch_bounds__(128)
gemm_wmma_k(const float* __restrict__ A, int lda,
            const __bf16* __restrict__ Wp,
            const float* __restrict__ bias,
            float* __restrict__ C, int ldc, int col_off,
            int M, int K, int act) {
  __shared__ alignas(16) __bf16 As[2][16 * 32];

  const int lane = threadIdx.x & 31;
  const int wave = threadIdx.x >> 5;
  const int ro   = blockIdx.x * 16;
  const int nblk0 = (blockIdx.y * 4 + wave) * 2;   // grid.y = N/128
  const int kblks = K >> 5;

  // staging map: 128 threads x float4 = 16x32 tile
  const int srow = threadIdx.x >> 3;
  const int scol = (threadIdx.x & 7) << 2;
  const int arow = (ro + srow < M) ? (ro + srow) : (M - 1);
  const float* ag = A + (size_t)arow * lda + scol;

  // A-fragment LDS source offsets for this lane
  const int r  = lane & 15;
  const int kh = (lane >> 4) << 3;

  const __bf16* wpt0 = Wp + (size_t)nblk0 * kblks * 512 + lane * 16;
  const __bf16* wpt1 = wpt0 + (size_t)kblks * 512;

  f32x8 acc0 = {}, acc1 = {};

  // prologue: stage k-block 0 into buffer 0
  {
    float4 v = *(const float4*)ag;
    cvt4_bf16(&As[0][srow * 32 + scol], v);
  }

  for (int kb = 0; kb < kblks; ++kb) {
    __syncthreads();   // staged buf[kb&1] visible; prior reads of buf[(kb+1)&1] done
    if (kb + 1 < kblks) {
      float4 v = *(const float4*)(ag + (kb + 1) * 32);
      cvt4_bf16(&As[(kb + 1) & 1][srow * 32 + scol], v);
      __builtin_prefetch(wpt0 + (size_t)(kb + 1) * 512, 0, 1);
      __builtin_prefetch(wpt1 + (size_t)(kb + 1) * 512, 0, 1);
    }

    const __bf16* ap0 = &As[kb & 1][r * 32 + kh];
    bf16x8 lo = *(const bf16x8*)ap0;
    bf16x8 hi = *(const bf16x8*)(ap0 + 16);
    bf16x16 a;
#pragma unroll
    for (int j = 0; j < 8; ++j) { a[j] = lo[j]; a[8 + j] = hi[j]; }

    bf16x16 b0 = *(const bf16x16*)(wpt0 + (size_t)kb * 512);
    bf16x16 b1 = *(const bf16x16*)(wpt1 + (size_t)kb * 512);

    acc0 = __builtin_amdgcn_wmma_f32_16x16x32_bf16(false, a, false, b0,
                                                   (short)0, acc0, false, false);
    acc1 = __builtin_amdgcn_wmma_f32_16x16x32_bf16(false, a, false, b1,
                                                   (short)0, acc1, false, false);
  }

  // epilogue: C/D layout: VGPR j -> M = j + (lane<16?0:8), N = lane&15
  const int nb0 = nblk0 * 16 + (lane & 15);
  const int nb1 = nb0 + 16;
  const float bv0 = bias ? bias[nb0] : 0.0f;
  const float bv1 = bias ? bias[nb1] : 0.0f;
  const int mbase = ro + ((lane >> 4) << 3);
#pragma unroll
  for (int j = 0; j < 8; ++j) {
    int m = mbase + j;
    if (m < M) {
      float v0 = acc0[j] + bv0;
      float v1 = acc1[j] + bv1;
      if (act == 1) { v0 = fmaxf(v0, 0.0f); v1 = fmaxf(v1, 0.0f); }
      float* cr = C + (size_t)m * ldc + col_off;
      cr[nb0] = v0;
      cr[nb1] = v1;
    }
  }
}

// ---------------------------------------------------------------------------
// Utility fill
// ---------------------------------------------------------------------------
__global__ void fill_k(float* p, float v, long long n) {
  long long i = (long long)blockIdx.x * 256 + threadIdx.x;
  if (i < n) p[i] = v;
}

// ---------------------------------------------------------------------------
// GENConv scatter-softmax aggregation (32 threads/edge, 4 channels/thread).
// Pass 1: per-(dst,channel) max of t*m via int atomicMax (tm > 0 so bitwise
// int compare is order-preserving; init 0 matches the empty->0 clamp and the
// softmax ratio is shift-invariant).
// ---------------------------------------------------------------------------
__global__ void edge_max_k(const int* __restrict__ src, const int* __restrict__ dstv,
                           const float* __restrict__ x, int ldx,
                           const float* __restrict__ tptr,
                           float* __restrict__ mx, int E) {
  int tid = blockIdx.x * 256 + threadIdx.x;
  int e = tid >> 5;
  if (e >= E) return;
  int c4 = (tid & 31) << 2;
  float t = *tptr;
  int s = src[e], d = dstv[e];
  float4 xv = *(const float4*)(x + (size_t)s * ldx + c4);
  int* mp = (int*)(mx + (size_t)d * 128 + c4);
  float xs[4] = {xv.x, xv.y, xv.z, xv.w};
#pragma unroll
  for (int j = 0; j < 4; ++j) {
    float m = fmaxf(xs[j], 0.0f) + 1e-7f;
    atomicMax(mp + j, __float_as_int(t * m));
  }
}

__global__ void edge_sum_k(const int* __restrict__ src, const int* __restrict__ dstv,
                           const float* __restrict__ x, int ldx,
                           const float* __restrict__ tptr,
                           const float* __restrict__ mx,
                           float* __restrict__ den, float* __restrict__ num, int E) {
  int tid = blockIdx.x * 256 + threadIdx.x;
  int e = tid >> 5;
  if (e >= E) return;
  int c4 = (tid & 31) << 2;
  float t = *tptr;
  int s = src[e], d = dstv[e];
  float4 xv = *(const float4*)(x + (size_t)s * ldx + c4);
  const float* mp = mx  + (size_t)d * 128 + c4;
  float* dp       = den + (size_t)d * 128 + c4;
  float* np       = num + (size_t)d * 128 + c4;
  float xs[4] = {xv.x, xv.y, xv.z, xv.w};
#pragma unroll
  for (int j = 0; j < 4; ++j) {
    float m = fmaxf(xs[j], 0.0f) + 1e-7f;
    float a = expf(t * m - mp[j]);
    atomicAdd(dp + j, a);
    atomicAdd(np + j, a * m);
  }
}

// out = num/max(den,1e-16) + x   (root residual)
__global__ void node_combine_k(const float* __restrict__ num, const float* __restrict__ den,
                               const float* __restrict__ x, int ldx,
                               float* __restrict__ outp, int n) {
  int i = blockIdx.x * 256 + threadIdx.x;
  if (i >= n * 128) return;
  int row = i >> 7, c = i & 127;
  float agg = num[i] / fmaxf(den[i], 1e-16f);
  outp[i] = agg + x[(size_t)row * ldx + c];
}

// ---------------------------------------------------------------------------
// LayerNorm(256) + ReLU in place. One wave per row (wave32).
// ---------------------------------------------------------------------------
__global__ void __launch_bounds__(256)
ln_relu_256_k(float* __restrict__ h, const float* __restrict__ g,
              const float* __restrict__ be, int n) {
  int wave = threadIdx.x >> 5, lane = threadIdx.x & 31;
  int row = blockIdx.x * 8 + wave;
  if (row >= n) return;
  float* hr = h + (size_t)row * 256;
  float v[8], s = 0.f, sq = 0.f;
#pragma unroll
  for (int j = 0; j < 8; ++j) {
    v[j] = hr[lane + 32 * j];
    s += v[j]; sq += v[j] * v[j];
  }
#pragma unroll
  for (int m = 16; m; m >>= 1) { s += __shfl_xor(s, m, 32); sq += __shfl_xor(sq, m, 32); }
  float mean = s * (1.0f / 256.0f);
  float var  = sq * (1.0f / 256.0f) - mean * mean;
  float inv  = rsqrtf(var + 1e-5f);
#pragma unroll
  for (int j = 0; j < 8; ++j) {
    int c = lane + 32 * j;
    float y = (v[j] - mean) * inv * g[c] + be[c];
    hr[c] = fmaxf(y, 0.0f);
  }
}

// x_next = x_prev + relu(LN128(h2)*gn+bn)
__global__ void __launch_bounds__(256)
ln128_relu_add_k(const float* __restrict__ h2, const float* __restrict__ gn,
                 const float* __restrict__ bn,
                 const float* __restrict__ xprev, int ldp,
                 float* __restrict__ xnext, int ldn, int n) {
  int wave = threadIdx.x >> 5, lane = threadIdx.x & 31;
  int row = blockIdx.x * 8 + wave;
  if (row >= n) return;
  const float* hr = h2 + (size_t)row * 128;
  float v[4], s = 0.f, sq = 0.f;
#pragma unroll
  for (int j = 0; j < 4; ++j) {
    v[j] = hr[lane + 32 * j];
    s += v[j]; sq += v[j] * v[j];
  }
#pragma unroll
  for (int m = 16; m; m >>= 1) { s += __shfl_xor(s, m, 32); sq += __shfl_xor(sq, m, 32); }
  float mean = s * (1.0f / 128.0f);
  float var  = sq * (1.0f / 128.0f) - mean * mean;
  float inv  = rsqrtf(var + 1e-5f);
#pragma unroll
  for (int j = 0; j < 4; ++j) {
    int c = lane + 32 * j;
    float y = (v[j] - mean) * inv * gn[c] + bn[c];
    xnext[(size_t)row * ldn + c] = xprev[(size_t)row * ldp + c] + fmaxf(y, 0.0f);
  }
}

// A_i = sum_c tanh(ha)*sigmoid(hb)*cw[c] + cb  (one wave per row)
__global__ void __launch_bounds__(256)
att_score_k(const float* __restrict__ ha, const float* __restrict__ hb,
            const float* __restrict__ cw, const float* __restrict__ cb,
            float* __restrict__ Araw, int n) {
  int wave = threadIdx.x >> 5, lane = threadIdx.x & 31;
  int row = blockIdx.x * 8 + wave;
  if (row >= n) return;
  const float* pa = ha + (size_t)row * 512;
  const float* pb = hb + (size_t)row * 512;
  float acc = 0.f;
#pragma unroll
  for (int j = 0; j < 16; ++j) {
    int c = lane + 32 * j;
    float av = tanhf(pa[c]);
    float bv = 1.0f / (1.0f + expf(-pb[c]));
    acc += av * bv * cw[c];
  }
#pragma unroll
  for (int m = 16; m; m >>= 1) acc += __shfl_xor(acc, m, 32);
  if (lane == 0) Araw[row] = acc + cb[0];
}

__global__ void __launch_bounds__(1024)
softmax_stats_k(const float* __restrict__ Araw, float* __restrict__ stats, int n) {
  __shared__ float red[1024];
  int t = threadIdx.x;
  float mx = -3.4e38f;
  for (int i = t; i < n; i += 1024) mx = fmaxf(mx, Araw[i]);
  red[t] = mx; __syncthreads();
  for (int s = 512; s; s >>= 1) { if (t < s) red[t] = fmaxf(red[t], red[t + s]); __syncthreads(); }
  float gmx = red[0]; __syncthreads();
  float sum = 0.f;
  for (int i = t; i < n; i += 1024) sum += expf(Araw[i] - gmx);
  red[t] = sum; __syncthreads();
  for (int s = 512; s; s >>= 1) { if (t < s) red[t] += red[t + s]; __syncthreads(); }
  if (t == 0) { stats[0] = gmx; stats[1] = red[0]; }
}

__global__ void __launch_bounds__(512)
weighted_pool_k(const float* __restrict__ Araw, const float* __restrict__ stats,
                const float* __restrict__ H, float* __restrict__ hp, int n) {
  int c = threadIdx.x;
  float gmx = stats[0], den = stats[1];
  float acc = 0.f;
  for (int i = blockIdx.x; i < n; i += gridDim.x)
    acc += (expf(Araw[i] - gmx) / den) * H[(size_t)i * 512 + c];
  atomicAdd(hp + c, acc);
}

__global__ void __launch_bounds__(512)
mean_pool_k(const float* __restrict__ Ft, float* __restrict__ tf, int n) {
  int c = threadIdx.x;
  float acc = 0.f;
  for (int i = blockIdx.x; i < n; i += gridDim.x)
    acc += Ft[(size_t)i * 512 + c];
  atomicAdd(tf + c, acc * (1.0f / (float)n));
}

// out[512] = act(v[512] @ W[512,512] + b)  (single block)
__global__ void __launch_bounds__(512)
vec_mat_512_k(const float* __restrict__ v, const float* __restrict__ W,
              const float* __restrict__ b, float* __restrict__ out, int act) {
  __shared__ float vs[512];
  int t = threadIdx.x;
  vs[t] = v[t];
  __syncthreads();
  float acc = 0.f;
  for (int k = 0; k < 512; ++k) acc += vs[k] * W[(size_t)k * 512 + t];
  acc += b[t];
  if (act) acc = fmaxf(acc, 0.0f);
  out[t] = acc;
}

// logits/probs/argmax: comb=[hp2(512), tf(512)] @ cls_w[1024,4] + cls_b
__global__ void __launch_bounds__(128)
classifier_k(const float* __restrict__ hp2, const float* __restrict__ tf,
             const float* __restrict__ W, const float* __restrict__ b,
             float* __restrict__ out) {
  __shared__ float lg[4];
  int wave = threadIdx.x >> 5, lane = threadIdx.x & 31;
  float acc = 0.f;
  for (int i = 0; i < 32; ++i) {
    int k = lane + 32 * i;
    float v = (k < 512) ? hp2[k] : tf[k - 512];
    acc += v * W[(size_t)k * 4 + wave];
  }
#pragma unroll
  for (int m = 16; m; m >>= 1) acc += __shfl_xor(acc, m, 32);
  if (lane == 0) lg[wave] = acc + b[wave];
  __syncthreads();
  if (threadIdx.x == 0) {
    float mx = lg[0];
    for (int j = 1; j < 4; ++j) mx = fmaxf(mx, lg[j]);
    float e[4], s = 0.f;
    for (int j = 0; j < 4; ++j) { e[j] = expf(lg[j] - mx); s += e[j]; }
    int amax = 0;
    for (int j = 1; j < 4; ++j) if (lg[j] > lg[amax]) amax = j;
    for (int j = 0; j < 4; ++j) { out[j] = lg[j]; out[4 + j] = e[j] / s; }
    out[8] = (float)amax;
  }
}

__global__ void copy_cols_k(const float* __restrict__ s, int lds_,
                            float* __restrict__ d, int ldd, int n, int w) {
  int i = blockIdx.x * 256 + threadIdx.x;
  if (i >= n * w) return;
  int r = i / w, c = i % w;
  d[(size_t)r * ldd + c] = s[(size_t)r * lds_ + c];
}

// ---------------------------------------------------------------------------
// Host orchestration.
// Input flattening assumed = setup_inputs() recursive dict-insertion order:
//  0 x, 1 edge_index, 2 tx, 3 tedge_index,
//  patch: 4 fc_w, 5 fc_b, conv1{6 t,7 w1,8 b1,9 g1,10 be1,11 w2,12 b2},
//  L0{13 t,14 w1,15 b1,16 g1,17 be1,18 w2,19 b2,20 gn,21 bn},
//  L1{22..30}, 31 phi_w,32 phi_b,33 att_a_w,34 att_a_b,35 att_b_w,36 att_b_b,
//  37 att_c_w,38 att_c_b,39 rho_w,40 rho_b,41 cls_w,42 cls_b,
//  tissue: 43 fc_w,44 fc_b, conv1{45..51}, L0{52..60}, L1{61..69}
// ---------------------------------------------------------------------------
extern "C" void kernel_launch(void* const* d_in, const int* in_sizes, int n_in,
                              void* d_out, int out_size, void* d_ws, size_t ws_size,
                              hipStream_t stream) {
  (void)n_in; (void)out_size; (void)ws_size;
  auto P = [&](int i) { return (const float*)d_in[i]; };

  const int Np = in_sizes[0] / 1024;
  const int Ep = in_sizes[1] / 2;
  const int Nt = in_sizes[2] / 1024;
  const int Et = in_sizes[3] / 2;
  const int* ei_p = (const int*)d_in[1];
  const int* ei_t = (const int*)d_in[3];

  // ---- workspace carve ----
  __bf16* wpcur = (__bf16*)d_ws;                        // packed weights (< 4 MB)
  float* sm    = (float*)((char*)d_ws + (4u << 20));
  float* hp    = sm;            // 512
  float* hp2   = sm + 512;      // 512
  float* tf    = sm + 1024;     // 512
  float* stats = sm + 1536;     // 2
  float* Fp    = sm + 2048;                             // [Np,512] feature concat
  float* Hb    = Fp   + (size_t)Np * 512;               // [Np,512] head h
  float* TMPR  = Hb   + (size_t)Np * 512;               // conv scratch / HA alias
  float* AGGR  = TMPR + (size_t)Np * 512;               // agg scratch / HB alias
  float* tmp = TMPR;
  float* h1  = TMPR + (size_t)Np * 128;
  float* h2  = TMPR + (size_t)Np * 384;
  float* HA  = TMPR;
  float* HB  = AGGR;
  float* mxb = AGGR;
  float* den = AGGR + (size_t)Np * 128;
  float* num = AGGR + (size_t)Np * 256;
  // tissue buffers alias Hb region (patch head finished before tissue runs)
  float* xt0   = Hb;
  float* Ft    = xt0   + (size_t)Nt * 128;
  float* tmp_t = Ft    + (size_t)Nt * 512;
  float* h1_t  = tmp_t + (size_t)Nt * 128;
  float* h2_t  = h1_t  + (size_t)Nt * 256;
  float* mx_t  = h2_t  + (size_t)Nt * 128;
  float* den_t = mx_t  + (size_t)Nt * 128;
  float* num_t = den_t + (size_t)Nt * 128;

  auto fill = [&](float* p, float v, size_t n) {
    fill_k<<<(int)((n + 255) / 256), 256, 0, stream>>>(p, v, (long long)n);
  };
  auto pack = [&](int pi, int K, int N) {
    __bf16* dst = wpcur;
    int total = K * N;
    pack_weights_k<<<(total + 255) / 256, 256, 0, stream>>>(P(pi), dst, K, N);
    wpcur += total;
    return dst;
  };
  auto gemm = [&](const float* A, int lda, const __bf16* Wp, const float* bias,
                  float* C, int ldc, int col, int M, int K, int N, int act) {
    dim3 g((M + 15) / 16, N / 128);   // block tile = 16 x 128
    gemm_wmma_k<<<g, dim3(128), 0, stream>>>(A, lda, Wp, bias, C, ldc, col, M, K, act);
  };
  auto conv = [&](const float* xin, int ldx, const __bf16* wp1, const __bf16* wp2,
                  int ti, int b1i, int g1i, int be1i, int b2i,
                  const int* src, const int* dstv, int n, int e,
                  float* Cout, int ldc, int col,
                  float* tmpb, float* h1b, float* mx_, float* den_, float* num_) {
    fill(mx_, 0.0f, (size_t)n * 128);
    fill(den_, 0.0f, (size_t)n * 128);
    fill(num_, 0.0f, (size_t)n * 128);
    int eb = (e * 32 + 255) / 256;
    edge_max_k<<<eb, 256, 0, stream>>>(src, dstv, xin, ldx, P(ti), mx_, e);
    edge_sum_k<<<eb, 256, 0, stream>>>(src, dstv, xin, ldx, P(ti), mx_, den_, num_, e);
    node_combine_k<<<((n * 128) + 255) / 256, 256, 0, stream>>>(num_, den_, xin, ldx, tmpb, n);
    gemm(tmpb, 128, wp1, P(b1i), h1b, 256, 0, n, 128, 256, 0);
    ln_relu_256_k<<<(n + 7) / 8, 256, 0, stream>>>(h1b, P(g1i), P(be1i), n);
    gemm(h1b, 256, wp2, P(b2i), Cout, ldc, col, n, 256, 128, 0);
  };

  // ---- pack all GEMM weights to bf16 fragment layout ----
  __bf16* wfc_p  = pack(4, 1024, 128);
  __bf16* w1c1p  = pack(7, 128, 256);
  __bf16* w2c1p  = pack(11, 256, 128);
  __bf16* w1l0p  = pack(14, 128, 256);
  __bf16* w2l0p  = pack(18, 256, 128);
  __bf16* w1l1p  = pack(23, 128, 256);
  __bf16* w2l1p  = pack(27, 256, 128);
  __bf16* wphi   = pack(31, 512, 512);
  __bf16* watta  = pack(33, 512, 512);
  __bf16* wattb  = pack(35, 512, 512);
  __bf16* wfc_t  = pack(43, 1024, 128);
  __bf16* w1c1t  = pack(46, 128, 256);
  __bf16* w2c1t  = pack(50, 256, 128);
  __bf16* w1l0t  = pack(53, 128, 256);
  __bf16* w2l0t  = pack(57, 256, 128);
  __bf16* w1l1t  = pack(62, 128, 256);
  __bf16* w2l1t  = pack(66, 256, 128);

  // ================= patch stack =================
  // x0 = relu(x @ fc_w + fc_b) -> F[:,0:128]
  gemm(P(0), 1024, wfc_p, P(5), Fp, 512, 0, Np, 1024, 128, 1);
  // conv1 -> x1 = F[:,128:256]
  conv(Fp, 512, w1c1p, w2c1p, 6, 8, 9, 10, 12, ei_p, ei_p + Ep, Np, Ep,
       Fp, 512, 128, tmp, h1, mxb, den, num);
  // deep layer 0: x1 -> x2 = F[:,256:384]
  conv(Fp + 128, 512, w1l0p, w2l0p, 13, 15, 16, 17, 19, ei_p, ei_p + Ep, Np, Ep,
       h2, 128, 0, tmp, h1, mxb, den, num);
  ln128_relu_add_k<<<(Np + 7) / 8, 256, 0, stream>>>(h2, P(20), P(21),
                                                     Fp + 128, 512, Fp + 256, 512, Np);
  // deep layer 1: x2 -> x3 = F[:,384:512]
  conv(Fp + 256, 512, w1l1p, w2l1p, 22, 24, 25, 26, 28, ei_p, ei_p + Ep, Np, Ep,
       h2, 128, 0, tmp, h1, mxb, den, num);
  ln128_relu_add_k<<<(Np + 7) / 8, 256, 0, stream>>>(h2, P(29), P(30),
                                                     Fp + 256, 512, Fp + 384, 512, Np);

  // ================= patch head =================
  gemm(Fp, 512, wphi,  P(32), Hb, 512, 0, Np, 512, 512, 1);   // h = relu(F@phi+b)
  gemm(Hb, 512, watta, P(34), HA, 512, 0, Np, 512, 512, 0);   // pre-tanh
  gemm(Hb, 512, wattb, P(36), HB, 512, 0, Np, 512, 512, 0);   // pre-sigmoid
  float* Araw = (float*)d_out + 9;
  att_score_k<<<(Np + 7) / 8, 256, 0, stream>>>(HA, HB, P(37), P(38), Araw, Np);
  softmax_stats_k<<<1, 1024, 0, stream>>>(Araw, stats, Np);
  fill(hp, 0.0f, 512);
  weighted_pool_k<<<256, 512, 0, stream>>>(Araw, stats, Hb, hp, Np);
  vec_mat_512_k<<<1, 512, 0, stream>>>(hp, P(39), P(40), hp2, 1);  // rho + relu

  // ================= tissue stack (aliases Hb region) =================
  gemm(P(2), 1024, wfc_t, P(44), xt0, 128, 0, Nt, 1024, 128, 1);
  conv(xt0, 128, w1c1t, w2c1t, 45, 47, 48, 49, 51, ei_t, ei_t + Et, Nt, Et,
       Ft, 512, 128, tmp_t, h1_t, mx_t, den_t, num_t);           // x1 -> col 128
  conv(Ft + 128, 512, w1l0t, w2l0t, 52, 54, 55, 56, 58, ei_t, ei_t + Et, Nt, Et,
       h2_t, 128, 0, tmp_t, h1_t, mx_t, den_t, num_t);
  ln128_relu_add_k<<<(Nt + 7) / 8, 256, 0, stream>>>(h2_t, P(59), P(60),
                                                     Ft + 128, 512, Ft + 256, 512, Nt);
  conv(Ft + 256, 512, w1l1t, w2l1t, 61, 63, 64, 65, 67, ei_t, ei_t + Et, Nt, Et,
       h2_t, 128, 0, tmp_t, h1_t, mx_t, den_t, num_t);
  ln128_relu_add_k<<<(Nt + 7) / 8, 256, 0, stream>>>(h2_t, P(68), P(69),
                                                     Ft + 256, 512, Ft + 384, 512, Nt);
  // dup_first: feats[0] = x1 too
  copy_cols_k<<<((Nt * 128) + 255) / 256, 256, 0, stream>>>(Ft + 128, 512, Ft, 512, Nt, 128);
  fill(tf, 0.0f, 512);
  mean_pool_k<<<64, 512, 0, stream>>>(Ft, tf, Nt);

  // ================= classifier =================
  classifier_k<<<1, 128, 0, stream>>>(hp2, tf, P(41), P(42), (float*)d_out);
}
